// LSTMCell_49039936586195
// MI455X (gfx1250) — matrix operand order repaired and checked by reading, not compiled, not run on hardware
//
#include <hip/hip_runtime.h>

// ---------------------------------------------------------------------------
// LSTM cell for MI455X (gfx1250, wave32, WMMA).
//
// gates = [x|h] @ [Wf|Wi|Wo|Wu] + b   : [8192x2048] @ [2048x4096], 137 GFLOP
// Compute-bound (714 FLOP/byte). BF16 WMMA (16x16x32, f32 accum) is 8x the
// per-issue throughput of FP32 WMMA (16x16x4) -> ~27us vs ~220us; memory
// floor is only ~8us @ 23.3 TB/s. Packed bf16 operands (48 MB) are L2
// resident (192 MB L2), so the GEMM streams from L2, not HBM.
//
// Staging uses gfx1250's GLOBAL_LOAD_ASYNC_TO_LDS_B128 (ASYNCcnt-tracked,
// no VGPR round trip); register-staged path kept as fallback.
// ---------------------------------------------------------------------------

typedef __bf16 v16bf __attribute__((ext_vector_type(16)));
typedef float  v8f   __attribute__((ext_vector_type(8)));
typedef int    v4i   __attribute__((ext_vector_type(4)));

typedef __attribute__((address_space(1))) v4i* gptr_v4i;   // global int4*
typedef __attribute__((address_space(3))) v4i* lptr_v4i;   // LDS int4*

#define BATCH    8192
#define DHID     1024
#define DIN      2048          // d_embed + d_hidden
#define KTILES   64            // DIN / 32
#define MTILES   512           // BATCH / 16
#define HTILES   64            // DHID / 16
#define A_PACK_BYTES ((size_t)BATCH * DIN * 2)       // 32 MiB bf16 A, frag order
#define B_PACK_BYTES ((size_t)DIN * 4 * DHID * 2)    // 16 MiB bf16 B, frag order

#define HAS_ASYNC_LDS __has_builtin(__builtin_amdgcn_global_load_async_to_lds_b128)

__device__ __forceinline__ void wait_async_zero() {
#if __has_builtin(__builtin_amdgcn_s_wait_asynccnt)
    __builtin_amdgcn_s_wait_asynccnt(0);
#else
    asm volatile("s_wait_asynccnt 0x0" ::: "memory");
#endif
}

// ---------------------------------------------------------------------------
// Fragment-order packing.
//
// A operand (16x32 bf16, ISA 7.12.2): lane = m%16 + 16*((k>>3)&1),
//   byte-in-lane = 16*(k>>4) + 2*(k&7).   (lanes 0-15: K {0..7,16..23})
// B operand (32x16 bf16, same k order as A): lane = n%16 + 16*((k>>3)&1),
//   byte-in-lane = 16*(k>>4) + 2*(k&7).              (lanes 0-15: K {0..7,16..23})
// Each 16x32 tile = 32 lanes x 32 B = 1 KiB, lane-major, so the GEMM reads
// exactly two ds_load_b128 per lane per fragment.
// Packed A layout: [tk][tm][1KiB frag]; packed B: [tk][gate][th][1KiB frag].
// ---------------------------------------------------------------------------

union Cvt8 { __bf16 b[8]; uint4 u; };

__global__ __launch_bounds__(256) void pack_a(const float* __restrict__ x,
                                              const float* __restrict__ h,
                                              unsigned char* __restrict__ aPack) {
    const int tid = blockIdx.x * 256 + threadIdx.x;   // 0 .. 2M-1
    const int c   = tid & 63;                         // 16B chunk within frag
    const int tk  = (tid >> 6) & 63;
    const int tm  = tid >> 12;                        // 0..511
    const int fl  = c >> 1;                           // frag lane 0..31
    const int hi  = c & 1;                            // which 16B half of lane
    const int mm    = fl & 15;
    const int khalf = fl >> 4;
    const int kk8 = hi * 16 + khalf * 8;              // A-layout K group of 8
    const int m   = tm * 16 + mm;
    const int kg  = tk * 32 + kk8;                    // global k in [0,2048)

    const float* src = (kg < 1024) ? (x + (size_t)m * 1024 + kg)
                                   : (h + (size_t)m * 1024 + (kg - 1024));
    const float4 f0 = ((const float4*)src)[0];
    const float4 f1 = ((const float4*)src)[1];
    Cvt8 cv;
    cv.b[0] = (__bf16)f0.x; cv.b[1] = (__bf16)f0.y;
    cv.b[2] = (__bf16)f0.z; cv.b[3] = (__bf16)f0.w;
    cv.b[4] = (__bf16)f1.x; cv.b[5] = (__bf16)f1.y;
    cv.b[6] = (__bf16)f1.z; cv.b[7] = (__bf16)f1.w;
    ((uint4*)(aPack + (((size_t)tk * MTILES + tm) << 10)))[c] = cv.u;
}

__global__ __launch_bounds__(256) void pack_b(const float* __restrict__ Wf,
                                              const float* __restrict__ Wi,
                                              const float* __restrict__ Wo,
                                              const float* __restrict__ Wu,
                                              unsigned char* __restrict__ bPack) {
    const int tid = blockIdx.x * 256 + threadIdx.x;   // 0 .. 1M-1
    const int c   = tid & 63;
    const int tk  = (tid >> 6) & 63;
    const int th  = (tid >> 12) & 63;
    const int g   = tid >> 18;                        // gate 0..3 (f,i,o,u)
    const int fl  = c >> 1;
    const int hi  = c & 1;
    const int nn     = fl & 15;
    const int khalf2 = fl >> 4;
    const int kk8 = hi * 16 + khalf2 * 8;             // B-layout K group of 8
    const int n   = th * 16 + nn;
    const int k0  = tk * 32 + kk8;

    const float* Wg = (g == 0) ? Wf : (g == 1) ? Wi : (g == 2) ? Wo : Wu;
    Cvt8 cv;
#pragma unroll
    for (int j = 0; j < 8; ++j)
        cv.b[j] = (__bf16)Wg[(size_t)(k0 + j) * DHID + n];
    ((uint4*)(bPack + ((((size_t)tk * 4 + g) * HTILES + th) << 10)))[c] = cv.u;
}

// ---------------------------------------------------------------------------
// Fused GEMM + LSTM epilogue.
// Block: 256 threads (8 waves), tile = 128 M x 64 hidden x 4 gates.
// Waves: 2 (M) x 4 (hidden); each wave: 4 m-tiles x 1 h-tile x 4 gates
// -> 16 v8f accumulators. K loop: 64 steps of 32, double-buffered LDS.
// ---------------------------------------------------------------------------

__device__ __forceinline__ float fsig(float x)  { return 1.0f / (1.0f + __expf(-x)); }
__device__ __forceinline__ float ftanh(float x) { return 1.0f - 2.0f / (__expf(2.0f * x) + 1.0f); }

__global__ __launch_bounds__(256) void lstm_wmma(
    const unsigned char* __restrict__ aPack,
    const unsigned char* __restrict__ bPack,
    const float* __restrict__ bF, const float* __restrict__ bI,
    const float* __restrict__ bO, const float* __restrict__ bU,
    const float* __restrict__ cIn, float* __restrict__ out) {
    __shared__ __align__(32) unsigned char lds[2][24576];  // 8KB A + 16KB B, x2

    const int t    = threadIdx.x;
    const int lane = t & 31;
    const int wave = t >> 5;
    const int wm   = wave >> 2;          // 0..1  (64-row group)
    const int wh   = wave & 3;           // 0..3  (16-hidden-col group)
    const int tm0  = blockIdx.x * 8;     // first m-tile of block
    const int th0  = blockIdx.y * 4;     // first h-tile of block

    v8f acc[4][4] = {};                  // [m-tile][gate]

#if HAS_ASYNC_LDS
    // gfx1250 async L2->LDS copy: no VGPR staging, tracked by ASYNCcnt.
    auto issuek = [&](int tk, int buf) {
        unsigned char* gA = const_cast<unsigned char*>(
            aPack + (((size_t)tk * MTILES + tm0) << 10)) + t * 16;
        unsigned char* gB = const_cast<unsigned char*>(
            bPack + (((size_t)tk * 4 * HTILES + th0) << 10)) + t * 16;
        unsigned char* lA = &lds[buf][0] + t * 16;
        unsigned char* lB = &lds[buf][8192] + t * 16;
        __builtin_amdgcn_global_load_async_to_lds_b128((gptr_v4i)(gA),          (lptr_v4i)(lA),         0, 0);
        __builtin_amdgcn_global_load_async_to_lds_b128((gptr_v4i)(gA + 4096),   (lptr_v4i)(lA + 4096),  0, 0);
        __builtin_amdgcn_global_load_async_to_lds_b128((gptr_v4i)(gB),          (lptr_v4i)(lB),         0, 0);
        __builtin_amdgcn_global_load_async_to_lds_b128((gptr_v4i)(gB + 65536),  (lptr_v4i)(lB + 4096),  0, 0);
        __builtin_amdgcn_global_load_async_to_lds_b128((gptr_v4i)(gB + 131072), (lptr_v4i)(lB + 8192),  0, 0);
        __builtin_amdgcn_global_load_async_to_lds_b128((gptr_v4i)(gB + 196608), (lptr_v4i)(lB + 12288), 0, 0);
    };
    issuek(0, 0);
    wait_async_zero();
    __syncthreads();
#else
    uint4 ra0, ra1, rb0, rb1, rb2, rb3;  // global->VGPR staging (24KB/block-step)
    auto loadk = [&](int tk) {
        const uint4* ga = (const uint4*)(aPack + (((size_t)tk * MTILES + tm0) << 10));
        ra0 = ga[t]; ra1 = ga[t + 256];
        const uint4* gb = (const uint4*)(bPack + (((size_t)tk * 4 * HTILES + th0) << 10));
        rb0 = gb[t]; rb1 = gb[t + 4096]; rb2 = gb[t + 8192]; rb3 = gb[t + 12288];
    };
    auto storek = [&](int buf) {
        uint4* la = (uint4*)(&lds[buf][0]);
        la[t] = ra0; la[t + 256] = ra1;
        uint4* lb = (uint4*)(&lds[buf][8192]);
        lb[t] = rb0; lb[t + 256] = rb1; lb[t + 512] = rb2; lb[t + 768] = rb3;
    };
    loadk(0);
    storek(0);
    __syncthreads();
#endif

    for (int tk = 0; tk < KTILES; ++tk) {
        const int cur = tk & 1;
#if HAS_ASYNC_LDS
        if (tk + 1 < KTILES) issuek(tk + 1, cur ^ 1);  // overlaps with WMMA below
#else
        if (tk + 1 < KTILES) loadk(tk + 1);
#endif

        const unsigned char* base = lds[cur];
        v16bf afrag[4];
#pragma unroll
        for (int mi = 0; mi < 4; ++mi)
            afrag[mi] = *(const v16bf*)(base + (((wm << 2) + mi) << 10) + (lane << 5));
#pragma unroll
        for (int g = 0; g < 4; ++g) {
            v16bf bfrag = *(const v16bf*)(base + 8192 + (((g << 2) + wh) << 10) + (lane << 5));
#pragma unroll
            for (int mi = 0; mi < 4; ++mi)
                acc[mi][g] = __builtin_amdgcn_wmma_f32_16x16x32_bf16(
                    false, afrag[mi], false, bfrag, (short)0, acc[mi][g], false, false);
        }

#if HAS_ASYNC_LDS
        if (tk + 1 < KTILES) wait_async_zero();        // this wave's copies landed
#else
        if (tk + 1 < KTILES) storek(cur ^ 1);
#endif
        __syncthreads();                               // everyone's copies visible
    }

    // ---- fused LSTM epilogue: this wave owns all 4 gates for its (r, col) ----
    const int colh = (blockIdx.y << 6) + (wh << 4) + (lane & 15);
    const float bfv = bF[colh], biv = bI[colh], bov = bO[colh], buv = bU[colh];
#pragma unroll
    for (int mi = 0; mi < 4; ++mi) {
        const int rbase = (blockIdx.x << 7) + (wm << 6) + (mi << 4) + ((lane >> 4) << 3);
#pragma unroll
        for (int v = 0; v < 8; ++v) {
            const int r = rbase + v;
            const size_t idx = (size_t)r * DHID + colh;
            const float f = fsig(acc[mi][0][v] + bfv);
            const float i = fsig(acc[mi][1][v] + biv);
            const float o = fsig(acc[mi][2][v] + bov);
            const float u = ftanh(acc[mi][3][v] + buv);
            const float nc = f * cIn[idx] + i * u;
            const float nh = o * ftanh(nc);
            out[idx] = nh;                                   // new_h
            out[(size_t)BATCH * DHID + idx] = nc;            // new_c
        }
    }
}

// ---------------------------------------------------------------------------
extern "C" void kernel_launch(void* const* d_in, const int* in_sizes, int n_in,
                              void* d_out, int out_size, void* d_ws, size_t ws_size,
                              hipStream_t stream) {
    const float* x  = (const float*)d_in[0];
    const float* h  = (const float*)d_in[1];
    const float* c  = (const float*)d_in[2];
    const float* Wf = (const float*)d_in[3]; const float* bf = (const float*)d_in[4];
    const float* Wi = (const float*)d_in[5]; const float* bi = (const float*)d_in[6];
    const float* Wo = (const float*)d_in[7]; const float* bo = (const float*)d_in[8];
    const float* Wu = (const float*)d_in[9]; const float* bu = (const float*)d_in[10];
    float* out = (float*)d_out;

    // Workspace: 48 MiB of packed bf16 operands (assumes ws_size >= 48 MiB).
    unsigned char* aPack = (unsigned char*)d_ws;
    unsigned char* bPack = aPack + A_PACK_BYTES;

    pack_a<<<(MTILES * KTILES * 64) / 256, 256, 0, stream>>>(x, h, aPack);
    pack_b<<<(4 * HTILES * KTILES * 64) / 256, 256, 0, stream>>>(Wf, Wi, Wo, Wu, bPack);

    dim3 grid(BATCH / 128, DHID / 64);
    lstm_wmma<<<grid, 256, 0, stream>>>(aPack, bPack, bf, bi, bo, bu, c, out);
}
